// GraphToGraph_1236950581835
// MI455X (gfx1250) — matrix operand order repaired and checked by reading, not compiled, not run on hardware
//
#include <hip/hip_runtime.h>
#include <hip/hip_bf16.h>

#define NODES 12288
#define EDGES 393216
#define DIN   128
#define DHID  128
#define DOUT  64

typedef __attribute__((ext_vector_type(16))) _Float16 v16h;
typedef __attribute__((ext_vector_type(8)))  _Float16 v8h;
typedef __attribute__((ext_vector_type(8)))  float    v8f;

// ---- init: deg=1 (self loop), agg buffers = 0 ---------------------------
__global__ void init_ws_kernel(float* __restrict__ deg, float* __restrict__ agg1,
                               float* __restrict__ agg2) {
  int idx = blockIdx.x * blockDim.x + threadIdx.x;
  if (idx < NODES)        deg[idx]  = 1.0f;
  if (idx < NODES * DHID) agg1[idx] = 0.0f;
  if (idx < NODES * DOUT) agg2[idx] = 0.0f;
}

// ---- in-degree over col (augmented with self loops via deg init) --------
__global__ void degree_kernel(const int* __restrict__ ei, float* __restrict__ deg) {
  int e = blockIdx.x * blockDim.x + threadIdx.x;
  if (e < EDGES) atomicAdd(&deg[ei[EDGES + e]], 1.0f);
}

__global__ void dinv_kernel(const float* __restrict__ deg, float* __restrict__ dinv) {
  int i = blockIdx.x * blockDim.x + threadIdx.x;
  if (i < NODES) dinv[i] = rsqrtf(fmaxf(deg[i], 1e-12f));
}

// ---- dense per-node GEMM: Y[i,:] = X[i,:] @ W ; one block per node ------
template<int FIN, int FOUT>
__global__ void gemm_node_kernel(const float* __restrict__ X, const float* __restrict__ W,
                                 float* __restrict__ Y) {
  __shared__ float xs[FIN];
  int i = blockIdx.x;
  for (int k = threadIdx.x; k < FIN; k += blockDim.x) xs[k] = X[(size_t)i * FIN + k];
  __syncthreads();
  int f = threadIdx.x;  // blockDim.x == FOUT
  float acc = 0.0f;
#pragma unroll 8
  for (int k = 0; k < FIN; ++k) acc = fmaf(xs[k], W[k * FOUT + f], acc);
  Y[(size_t)i * FOUT + f] = acc;
}

// ---- edge scatter: agg[c,:] += dinv[r]*dinv[c]*H[r,:] -------------------
template<int F, int V>
__global__ void scatter_kernel(const int* __restrict__ ei, const float* __restrict__ dinv,
                               const float* __restrict__ H, float* __restrict__ agg) {
  const int perEdge = F / V;
  long long idx = (long long)blockIdx.x * blockDim.x + threadIdx.x;
  long long e = idx / perEdge;
  int f = (int)(idx % perEdge) * V;
  if (e >= EDGES) return;
  int r = ei[e];
  int c = ei[EDGES + e];
  float nrm = dinv[r] * dinv[c];
  const float* hr = H + (size_t)r * F + f;
  float* ac = agg + (size_t)c * F + f;
#pragma unroll
  for (int t = 0; t < V; ++t) atomicAdd(&ac[t], nrm * hr[t]);
}

// ---- layer-1 epilogue: h1 = relu(agg1 + dinv^2*H1 + b1) -----------------
__global__ void finish1_kernel(const float* __restrict__ agg, const float* __restrict__ H,
                               const float* __restrict__ dinv, const float* __restrict__ b,
                               float* __restrict__ out) {
  int idx = blockIdx.x * blockDim.x + threadIdx.x;
  if (idx >= NODES * DHID) return;
  int i = idx / DHID, f = idx % DHID;
  float v = agg[idx] + dinv[i] * dinv[i] * H[idx] + b[f];
  out[idx] = fmaxf(v, 0.0f);
}

// ---- layer-2 epilogue: latent -> f16 for the WMMA gram ------------------
__global__ void finish2_kernel(const float* __restrict__ agg, const float* __restrict__ H,
                               const float* __restrict__ dinv, const float* __restrict__ b,
                               _Float16* __restrict__ Lh) {
  int idx = blockIdx.x * blockDim.x + threadIdx.x;
  if (idx >= NODES * DOUT) return;
  int i = idx / DOUT, f = idx % DOUT;
  float v = agg[idx] + dinv[i] * dinv[i] * H[idx] + b[f];
  Lh[idx] = (_Float16)v;
}

// ---- gram: out = L @ L^T via v_wmma_f32_16x16x32_f16, K=64 (2 WMMAs) ----
// 8 waves/block; wave w computes tile (blockIdx.y, blockIdx.x*8+w).
__global__ void __launch_bounds__(256) gram_wmma_kernel(const _Float16* __restrict__ L,
                                                        float* __restrict__ out) {
  int lane = threadIdx.x & 31;
  int wave = threadIdx.x >> 5;
  int half = lane >> 4;   // lane group (0: lanes 0-15, 1: lanes 16-31)
  int mrow = lane & 15;

  int iBase = blockIdx.y * 16;
  int jBase = (blockIdx.x * 8 + wave) * 16;

  const _Float16* arow = L + (size_t)(iBase + mrow) * DOUT;  // A row (M = mrow)
  const _Float16* brow = L + (size_t)(jBase + mrow) * DOUT;  // B col (N = mrow), B[k][n]=L[jBase+n][k]

  v8f c = {};
#pragma unroll
  for (int kk = 0; kk < 2; ++kk) {
    int k0 = kk * 32;
    // A (16x32 MxK): lanes 0-15 hold K k0..k0+15, lanes 16-31 hold K k0+16..k0+31
    v16h a = *(const v16h*)(arow + k0 + half * 16);
    // B (32x16 KxN): lanes 0-15 hold K k0..k0+15, lanes 16-31 hold K k0+16..k0+31
    v16h b = *(const v16h*)(brow + k0 + half * 16);
    c = __builtin_amdgcn_wmma_f32_16x16x32_f16(false, a, false, b, (short)0, c, false, false);
  }
  // C/D: VGPR v -> row iBase + half*8 + v, col jBase + mrow
#pragma unroll
  for (int v = 0; v < 8; ++v) {
    out[(size_t)(iBase + half * 8 + v) * NODES + (jBase + mrow)] = c[v];
  }
}

extern "C" void kernel_launch(void* const* d_in, const int* in_sizes, int n_in,
                              void* d_out, int out_size, void* d_ws, size_t ws_size,
                              hipStream_t stream) {
  const float* x  = (const float*)d_in[0];
  const int*   ei = (const int*)d_in[1];   // [2, E] row-major (JAX default x64-off -> int32)
  const float* W1 = (const float*)d_in[2];
  const float* b1 = (const float*)d_in[3];
  const float* W2 = (const float*)d_in[4];
  const float* b2 = (const float*)d_in[5];
  float* out = (float*)d_out;

  // workspace carve-up (floats); total ~27 MB
  float* deg  = (float*)d_ws;                         // N
  float* dinv = deg  + NODES;                         // N
  float* H1   = dinv + NODES;                         // N*128
  float* agg1 = H1   + (size_t)NODES * DHID;          // N*128
  float* h1   = agg1 + (size_t)NODES * DHID;          // N*128
  float* H2   = h1   + (size_t)NODES * DHID;          // N*64
  float* agg2 = H2   + (size_t)NODES * DOUT;          // N*64
  _Float16* Lh = (_Float16*)(agg2 + (size_t)NODES * DOUT);  // N*64 halves

  init_ws_kernel<<<(NODES * DHID + 255) / 256, 256, 0, stream>>>(deg, agg1, agg2);
  degree_kernel<<<(EDGES + 255) / 256, 256, 0, stream>>>(ei, deg);
  dinv_kernel<<<(NODES + 255) / 256, 256, 0, stream>>>(deg, dinv);

  gemm_node_kernel<DIN, DHID><<<NODES, DHID, 0, stream>>>(x, W1, H1);
  scatter_kernel<DHID, 4><<<(EDGES * (DHID / 4) + 255) / 256, 256, 0, stream>>>(ei, dinv, H1, agg1);
  finish1_kernel<<<(NODES * DHID + 255) / 256, 256, 0, stream>>>(agg1, H1, dinv, b1, h1);

  gemm_node_kernel<DHID, DOUT><<<NODES, DOUT, 0, stream>>>(h1, W2, H2);
  scatter_kernel<DOUT, 4><<<(EDGES * (DOUT / 4) + 255) / 256, 256, 0, stream>>>(ei, dinv, H2, agg2);
  finish2_kernel<<<(NODES * DOUT + 255) / 256, 256, 0, stream>>>(agg2, H2, dinv, b2, Lh);

  dim3 ggrid(NODES / 16 / 8, NODES / 16);  // (96, 768): 8 tiles per block in j
  gram_wmma_kernel<<<ggrid, 256, 0, stream>>>(Lh, out);
}
